// Diff_MultiHeadAttention_7670811591181
// MI455X (gfx1250) — matrix operand order repaired and checked by reading, not compiled, not run on hardware
//
#include <hip/hip_runtime.h>
#include <hip/hip_bf16.h>
#include <stdint.h>

// Problem sizes (fixed by the reference)
#define NTOK 4096
#define HIDD 256
#define PROJ 2048

using bf16 = __bf16;
typedef bf16  v8bf  __attribute__((ext_vector_type(8)));
typedef bf16  v16bf __attribute__((ext_vector_type(16)));
typedef float v8f   __attribute__((ext_vector_type(8)));

// ---------- helpers ----------------------------------------------------------

__device__ __forceinline__ bf16 f2bf(float f) {
    union { float f; uint32_t u; } v{f};
    uint32_t r = v.u + 0x7FFFu + ((v.u >> 16) & 1u);   // RNE
    union { uint16_t u; bf16 b; } o{(uint16_t)(r >> 16)};
    return o.b;
}

__device__ __forceinline__ v8f wmma_bf16(v16bf a, v16bf b, v8f c) {
    // (neg_a, A, neg_b, B, c_mod, C, reuse_a, reuse_b)
    return __builtin_amdgcn_wmma_f32_16x16x32_bf16(false, a, false, b, (short)0, c,
                                                   false, false);
}

// A-operand (16x32 bf16): lane m = lane&15, kg = lane>>4.
// elements 0..7  -> K = k0 + kg*16 + e          (contiguous, 16B)
// elements 8..15 -> K = k0 + kg*16 + 8 + (e-8)  (contiguous, 16B)
__device__ __forceinline__ v16bf load_a16(const bf16* row, int k0, int kg) {
    v8bf lo = *reinterpret_cast<const v8bf*>(row + k0 + 16 * kg);
    v8bf hi = *reinterpret_cast<const v8bf*>(row + k0 + 8 + 16 * kg);
    v16bf a;
#pragma unroll
    for (int e = 0; e < 8; ++e) { a[e] = lo[e]; a[e + 8] = hi[e]; }
    return a;
}

// B-operand (32x16 bf16): lane n = lane&15, kg = lane>>4; per-lane elements are
// K = k0 + kg*16 + e, fully contiguous -> single 32B load from a K-major row.
__device__ __forceinline__ v16bf load_b16(const bf16* nrow, int k0, int kg) {
    return *reinterpret_cast<const v16bf*>(nrow + k0 + kg * 16);
}

// Two 16x16 score tiles sharing one A operand, full K=PROJ contraction.
__device__ __forceinline__ void score_tile2(const bf16* __restrict__ arow,
                                            const bf16* __restrict__ brow0,
                                            const bf16* __restrict__ brow1,
                                            int kg, v8f& c0, v8f& c1) {
#pragma unroll 4
    for (int k0 = 0; k0 < PROJ; k0 += 32) {
        v16bf a = load_a16(arow, k0, kg);
        c0 = wmma_bf16(a, load_b16(brow0, k0, kg), c0);
        c1 = wmma_bf16(a, load_b16(brow1, k0, kg), c1);
    }
}

// ---------- tiny prep kernels -----------------------------------------------

__global__ void lam_kernel(const float* __restrict__ lq1, const float* __restrict__ lk1,
                           const float* __restrict__ lq2, const float* __restrict__ lk2,
                           const int* __restrict__ layer, float* __restrict__ scal) {
    __shared__ float s1[256], s2[256];
    int t = threadIdx.x;                      // HIDD == 256 exactly
    s1[t] = lq1[t] * lk1[t];
    s2[t] = lq2[t] * lk2[t];
    __syncthreads();
    for (int off = 128; off > 0; off >>= 1) {
        if (t < off) { s1[t] += s1[t + off]; s2[t] += s2[t + off]; }
        __syncthreads();
    }
    if (t == 0) {
        float lam1 = __expf(s1[0]);
        float lam2 = __expf(s2[0]);
        float li   = 0.8f - 0.6f * __expf(-0.3f * (float)layer[0]);
        scal[0] = lam1 - lam2 + li;   // lam_full
        scal[1] = 1.0f - li;          // (1 - lam_init)
    }
}

__global__ void cvt_rm(const float* __restrict__ src, bf16* __restrict__ dst, int n) {
    int i = blockIdx.x * 256 + threadIdx.x;
    if (i < n) dst[i] = f2bf(src[i]);
}

// f32 [K,Nc] row-major  ->  bf16 [Nc,K]  (K-contiguous rows for WMMA B-operand)
__global__ void cvt_T(const float* __restrict__ src, bf16* __restrict__ dst,
                      int K, int Nc) {
    int i = blockIdx.x * 256 + threadIdx.x;
    if (i < K * Nc) {
        int k = i / Nc, n = i % Nc;
        dst[(size_t)n * K + k] = f2bf(src[i]);
    }
}

// ---------- projection GEMM: C[M,PROJ] = A[M,HIDD] @ W + bias ---------------
// A bf16 row-major, BT bf16 [PROJ,HIDD] (K-contiguous). TRANS=1 stores C^T.
// 8 waves, each computes 2 adjacent 16x16 C tiles sharing one A operand.
// Block tile: 32 rows x 128 cols.
template <int TRANS>
__global__ void proj_gemm(const bf16* __restrict__ A, const bf16* __restrict__ BT,
                          const float* __restrict__ bias, bf16* __restrict__ C) {
    const int wave = threadIdx.x >> 5, lane = threadIdx.x & 31;
    const int m = lane & 15, n = lane & 15, kg = lane >> 4;
    const int row0 = blockIdx.x * 32 + (wave >> 2) * 16;    // grid.x = NTOK/32
    const int col0 = blockIdx.y * 128 + (wave & 3) * 32;    // grid.y = PROJ/128
    const bf16* arow  = A  + (size_t)(row0 + m) * HIDD;
    const bf16* brow0 = BT + (size_t)(col0 + n) * HIDD;
    const bf16* brow1 = BT + (size_t)(col0 + 16 + n) * HIDD;
    v8f c0 = {0.f, 0.f, 0.f, 0.f, 0.f, 0.f, 0.f, 0.f};
    v8f c1 = c0;
#pragma unroll
    for (int k0 = 0; k0 < HIDD; k0 += 32) {
        v16bf a = load_a16(arow, k0, kg);
        c0 = wmma_bf16(a, load_b16(brow0, k0, kg), c0);
        c1 = wmma_bf16(a, load_b16(brow1, k0, kg), c1);
    }
    const float bn0 = bias[col0 + n];
    const float bn1 = bias[col0 + 16 + n];
#pragma unroll
    for (int r = 0; r < 8; ++r) {
        int mm = row0 + r + kg * 8;
        bf16 v0 = f2bf(c0[r] + bn0);
        bf16 v1 = f2bf(c1[r] + bn1);
        if (TRANS) {                                        // vv^T layout
            C[(size_t)(col0 + n) * NTOK + mm]      = v0;
            C[(size_t)(col0 + 16 + n) * NTOK + mm] = v1;
        } else {
            C[(size_t)mm * PROJ + col0 + n]      = v0;
            C[(size_t)mm * PROJ + col0 + 16 + n] = v1;
        }
    }
}

// ---------- pass A: streaming row max / sum-exp for S1 and S2 ---------------
// Block = 16 query rows; key block BN=128. 8 waves: waves 0..3 -> S1 (2 j-tiles
// each), waves 4..7 -> S2. Online (m,l) state lives in LDS; updates are fully
// parallel: 32 groups of 8 threads, each thread reduces 16 elements, then a
// 3-step shfl_xor tree inside the wave.
__global__ void attn_stats(const bf16* __restrict__ q1, const bf16* __restrict__ k1,
                           const bf16* __restrict__ q2, const bf16* __restrict__ k2,
                           float* __restrict__ m1, float* __restrict__ l1,
                           float* __restrict__ m2, float* __restrict__ l2) {
    __shared__ __align__(32) float sL[2][16][128];
    __shared__ float rmS[2][16], rlS[2][16];
    const int tid = threadIdx.x, wave = tid >> 5, lane = tid & 31;
    const int i0 = blockIdx.x * 16;
    const int mat = wave >> 2, jp = wave & 3;
    const int m = lane & 15, n = lane & 15, kg = lane >> 4;
    const bf16* arow = (mat ? q2 : q1) + (size_t)(i0 + m) * PROJ;
    const bf16* Kp   =  mat ? k2 : k1;
    if (tid < 32) { rmS[tid >> 4][tid & 15] = -3.0e38f; rlS[tid >> 4][tid & 15] = 0.f; }
    __syncthreads();

    const int g = tid >> 3, sub = tid & 7;    // reduction group / sub-lane
    const int mt = g >> 4, row = g & 15;

    for (int j0 = 0; j0 < NTOK; j0 += 128) {
        const bf16* brow0 = Kp + (size_t)(j0 + jp * 32 + n) * PROJ;
        const bf16* brow1 = brow0 + (size_t)16 * PROJ;
        v8f c0 = {0.f, 0.f, 0.f, 0.f, 0.f, 0.f, 0.f, 0.f};
        v8f c1 = c0;
        score_tile2(arow, brow0, brow1, kg, c0, c1);
#pragma unroll
        for (int r = 0; r < 8; ++r) {
            sL[mat][r + kg * 8][jp * 32 + n]      = c0[r];
            sL[mat][r + kg * 8][jp * 32 + 16 + n] = c1[r];
        }
        __syncthreads();
        // parallel online (m,l) update: 16 elements/thread + shfl tree
        {
            const float* xr = &sL[mt][row][sub * 16];
            float bmax = -3.0e38f;
#pragma unroll
            for (int e = 0; e < 16; ++e) bmax = fmaxf(bmax, xr[e]);
            bmax = fmaxf(bmax, __shfl_xor(bmax, 1, 32));
            bmax = fmaxf(bmax, __shfl_xor(bmax, 2, 32));
            bmax = fmaxf(bmax, __shfl_xor(bmax, 4, 32));
            float rm_old = rmS[mt][row];
            float nm = fmaxf(rm_old, bmax);
            float s = 0.f;
#pragma unroll
            for (int e = 0; e < 16; ++e) s += __expf(xr[e] - nm);
            s += __shfl_xor(s, 1, 32);
            s += __shfl_xor(s, 2, 32);
            s += __shfl_xor(s, 4, 32);
            if (sub == 0) {
                rlS[mt][row] = rlS[mt][row] * __expf(rm_old - nm) + s;
                rmS[mt][row] = nm;
            }
        }
        __syncthreads();
    }
    if (tid < 32) {
        const int r2 = tid & 15, m2f = tid >> 4;
        (m2f ? m2 : m1)[i0 + r2] = rmS[m2f][r2];
        (m2f ? l2 : l1)[i0 + r2] = rlS[m2f][r2];
    }
}

// ---------- pass B: recompute scores, form diff-probabilities, PV GEMM ------
// Key block BN=128. acc: each wave owns 16 rows x 256 PROJ-cols (16 C tiles).
__global__ void attn_pv(const bf16* __restrict__ q1, const bf16* __restrict__ k1,
                        const bf16* __restrict__ q2, const bf16* __restrict__ k2,
                        const bf16* __restrict__ vvT,
                        const float* __restrict__ m1, const float* __restrict__ l1,
                        const float* __restrict__ m2, const float* __restrict__ l2,
                        const float* __restrict__ scal, float* __restrict__ attn) {
    __shared__ __align__(32) float sL[2][16][128];
    __shared__ __align__(32) bf16  pL[16][128];
    __shared__ float sm1[16], si1[16], sm2[16], si2[16];
    const int tid = threadIdx.x, wave = tid >> 5, lane = tid & 31;
    const int i0 = blockIdx.x * 16;
    const int mat = wave >> 2, jp = wave & 3;
    const int m = lane & 15, n = lane & 15, kg = lane >> 4;
    const float lam = scal[0];
    const bf16* arow = (mat ? q2 : q1) + (size_t)(i0 + m) * PROJ;
    const bf16* Kp   =  mat ? k2 : k1;
    if (tid < 16) {
        sm1[tid] = m1[i0 + tid]; si1[tid] = 1.f / l1[i0 + tid];
        sm2[tid] = m2[i0 + tid]; si2[tid] = 1.f / l2[i0 + tid];
    }
    __syncthreads();

    const int colbase = wave * 256;           // this wave's PROJ slice
    v8f z = {0.f, 0.f, 0.f, 0.f, 0.f, 0.f, 0.f, 0.f};
    v8f acc[16];
#pragma unroll
    for (int t = 0; t < 16; ++t) acc[t] = z;

    for (int j0 = 0; j0 < NTOK; j0 += 128) {
        // scores for this 128-key block (2 tiles per wave, shared A)
        const bf16* brow0 = Kp + (size_t)(j0 + jp * 32 + n) * PROJ;
        const bf16* brow1 = brow0 + (size_t)16 * PROJ;
        v8f c0 = z, c1 = z;
        score_tile2(arow, brow0, brow1, kg, c0, c1);
#pragma unroll
        for (int r = 0; r < 8; ++r) {
            sL[mat][r + kg * 8][jp * 32 + n]      = c0[r];
            sL[mat][r + kg * 8][jp * 32 + 16 + n] = c1[r];
        }
        __syncthreads();
        // p = exp(s1-m1)/l1 - lam * exp(s2-m2)/l2   (bf16 PV A-operand)
#pragma unroll
        for (int idx = tid; idx < 16 * 128; idx += 256) {
            int rr = idx >> 7, jj = idx & 127;
            float e1 = __expf(sL[0][rr][jj] - sm1[rr]) * si1[rr];
            float e2 = __expf(sL[1][rr][jj] - sm2[rr]) * si2[rr];
            pL[rr][jj] = f2bf(e1 - lam * e2);
        }
        __syncthreads();
        // PV: A = pL (16x128, from LDS), B = vvT rows (K-contiguous over keys)
#pragma unroll
        for (int kk = 0; kk < 4; ++kk) {
            v16bf a = load_a16(&pL[m][0], kk * 32, kg);
#pragma unroll
            for (int t = 0; t < 16; ++t) {
                const bf16* vr = vvT + (size_t)(colbase + t * 16 + n) * NTOK
                                 + j0 + kk * 32;
                acc[t] = wmma_bf16(a, *reinterpret_cast<const v16bf*>(vr + kg * 16),
                                   acc[t]);
            }
        }
        __syncthreads();
    }
#pragma unroll
    for (int t = 0; t < 16; ++t)
#pragma unroll
        for (int r = 0; r < 8; ++r)
            attn[(size_t)(i0 + r + kg * 8) * PROJ + colbase + t * 16 + n] = acc[t][r];
}

// ---------- LayerNorm over PROJ, scaled by (1-lam_init), bf16 out -----------
__global__ void ln_kernel(const float* __restrict__ attn, const float* __restrict__ g,
                          const float* __restrict__ b, const float* __restrict__ scal,
                          bf16* __restrict__ out) {
    const int row = blockIdx.x, tid = threadIdx.x;
    const int wave = tid >> 5, lane = tid & 31;
    const float* x = attn + (size_t)row * PROJ;
    float s = 0.f, sq = 0.f;
    for (int c = tid; c < PROJ; c += 256) { float v = x[c]; s += v; sq += v * v; }
#pragma unroll
    for (int off = 16; off > 0; off >>= 1) {
        s  += __shfl_xor(s, off, 32);
        sq += __shfl_xor(sq, off, 32);
    }
    __shared__ float rs[8], rq[8];
    if (lane == 0) { rs[wave] = s; rq[wave] = sq; }
    __syncthreads();
    if (tid == 0) {
        float ts = 0.f, tq = 0.f;
        for (int i = 0; i < 8; ++i) { ts += rs[i]; tq += rq[i]; }
        rs[0] = ts; rq[0] = tq;
    }
    __syncthreads();
    const float mu  = rs[0] * (1.0f / PROJ);
    const float var = rq[0] * (1.0f / PROJ) - mu * mu;
    const float inv = rsqrtf(var + 1e-5f);
    const float om  = scal[1];
    for (int c = tid; c < PROJ; c += 256) {
        float y = (x[c] - mu) * inv;
        out[(size_t)row * PROJ + c] = f2bf((y * g[c] + b[c]) * om);
    }
}

// ---------- final GEMM: out[N,HIDD] = attn_n @ Wo + bo ----------------------
// 2 C tiles per wave sharing A; block tile 32 x 128.
__global__ void out_gemm(const bf16* __restrict__ A, const bf16* __restrict__ WoT,
                         const float* __restrict__ bo, float* __restrict__ out) {
    const int wave = threadIdx.x >> 5, lane = threadIdx.x & 31;
    const int m = lane & 15, n = lane & 15, kg = lane >> 4;
    const int row0 = blockIdx.x * 32 + (wave >> 2) * 16;    // grid.x = NTOK/32
    const int col0 = blockIdx.y * 128 + (wave & 3) * 32;    // grid.y = HIDD/128
    const bf16* arow  = A   + (size_t)(row0 + m) * PROJ;
    const bf16* brow0 = WoT + (size_t)(col0 + n) * PROJ;
    const bf16* brow1 = WoT + (size_t)(col0 + 16 + n) * PROJ;
    v8f c0 = {0.f, 0.f, 0.f, 0.f, 0.f, 0.f, 0.f, 0.f};
    v8f c1 = c0;
#pragma unroll 4
    for (int k0 = 0; k0 < PROJ; k0 += 32) {
        v16bf a = load_a16(arow, k0, kg);
        c0 = wmma_bf16(a, load_b16(brow0, k0, kg), c0);
        c1 = wmma_bf16(a, load_b16(brow1, k0, kg), c1);
    }
    const float bn0 = bo[col0 + n];
    const float bn1 = bo[col0 + 16 + n];
#pragma unroll
    for (int r = 0; r < 8; ++r) {
        out[(size_t)(row0 + r + kg * 8) * HIDD + col0 + n]      = c0[r] + bn0;
        out[(size_t)(row0 + r + kg * 8) * HIDD + col0 + 16 + n] = c1[r] + bn1;
    }
}

// ---------- host launcher ---------------------------------------------------

extern "C" void kernel_launch(void* const* d_in, const int* in_sizes, int n_in,
                              void* d_out, int out_size, void* d_ws, size_t ws_size,
                              hipStream_t stream) {
    (void)in_sizes; (void)n_in; (void)out_size; (void)ws_size;
    const float* q    = (const float*)d_in[0];
    const float* k    = (const float*)d_in[1];
    const float* v    = (const float*)d_in[2];
    const int*   layer= (const int*)  d_in[3];
    const float* Wq1  = (const float*)d_in[4];
    const float* bq1  = (const float*)d_in[5];
    const float* Wk1  = (const float*)d_in[6];
    const float* bk1  = (const float*)d_in[7];
    const float* Wq2  = (const float*)d_in[8];
    const float* bq2  = (const float*)d_in[9];
    const float* Wk2  = (const float*)d_in[10];
    const float* bk2  = (const float*)d_in[11];
    const float* Wv   = (const float*)d_in[12];
    const float* bv   = (const float*)d_in[13];
    const float* ln_g = (const float*)d_in[14];
    const float* ln_b = (const float*)d_in[15];
    const float* Wo   = (const float*)d_in[16];
    const float* bo   = (const float*)d_in[17];
    const float* lq1  = (const float*)d_in[18];
    const float* lk1  = (const float*)d_in[19];
    const float* lq2  = (const float*)d_in[20];
    const float* lk2  = (const float*)d_in[21];
    float* out = (float*)d_out;

    // workspace carve-up (~141 MB total)
    char* w = (char*)d_ws;
    size_t off = 0;
    auto alloc = [&](size_t bytes) -> void* {
        void* p = w + off;
        off = (off + bytes + 255) & ~(size_t)255;
        return p;
    };
    float* scal  = (float*)alloc(2 * sizeof(float));
    bf16*  Qb    = (bf16*)alloc((size_t)NTOK * HIDD * 2);
    bf16*  Kb    = (bf16*)alloc((size_t)NTOK * HIDD * 2);
    bf16*  Vb    = (bf16*)alloc((size_t)NTOK * HIDD * 2);
    bf16*  Wq1T  = (bf16*)alloc((size_t)PROJ * HIDD * 2);
    bf16*  Wk1T  = (bf16*)alloc((size_t)PROJ * HIDD * 2);
    bf16*  Wq2T  = (bf16*)alloc((size_t)PROJ * HIDD * 2);
    bf16*  Wk2T  = (bf16*)alloc((size_t)PROJ * HIDD * 2);
    bf16*  WvT   = (bf16*)alloc((size_t)PROJ * HIDD * 2);
    bf16*  WoT   = (bf16*)alloc((size_t)HIDD * PROJ * 2);
    bf16*  q1b   = (bf16*)alloc((size_t)NTOK * PROJ * 2);
    bf16*  k1b   = (bf16*)alloc((size_t)NTOK * PROJ * 2);
    bf16*  q2b   = (bf16*)alloc((size_t)NTOK * PROJ * 2);
    bf16*  k2b   = (bf16*)alloc((size_t)NTOK * PROJ * 2);
    bf16*  vvT   = (bf16*)alloc((size_t)PROJ * NTOK * 2);
    float* m1    = (float*)alloc((size_t)NTOK * 4);
    float* l1    = (float*)alloc((size_t)NTOK * 4);
    float* m2    = (float*)alloc((size_t)NTOK * 4);
    float* l2    = (float*)alloc((size_t)NTOK * 4);
    float* attnF = (float*)alloc((size_t)NTOK * PROJ * 4);
    bf16*  attnN = (bf16*)alloc((size_t)NTOK * PROJ * 2);

    // scalars + conversions
    lam_kernel<<<1, 256, 0, stream>>>(lq1, lk1, lq2, lk2, layer, scal);
    const int nQ = NTOK * HIDD;
    cvt_rm<<<(nQ + 255) / 256, 256, 0, stream>>>(q, Qb, nQ);
    cvt_rm<<<(nQ + 255) / 256, 256, 0, stream>>>(k, Kb, nQ);
    cvt_rm<<<(nQ + 255) / 256, 256, 0, stream>>>(v, Vb, nQ);
    const int nW = HIDD * PROJ;
    cvt_T<<<(nW + 255) / 256, 256, 0, stream>>>(Wq1, Wq1T, HIDD, PROJ);
    cvt_T<<<(nW + 255) / 256, 256, 0, stream>>>(Wk1, Wk1T, HIDD, PROJ);
    cvt_T<<<(nW + 255) / 256, 256, 0, stream>>>(Wq2, Wq2T, HIDD, PROJ);
    cvt_T<<<(nW + 255) / 256, 256, 0, stream>>>(Wk2, Wk2T, HIDD, PROJ);
    cvt_T<<<(nW + 255) / 256, 256, 0, stream>>>(Wv,  WvT,  HIDD, PROJ);
    cvt_T<<<(nW + 255) / 256, 256, 0, stream>>>(Wo,  WoT,  PROJ, HIDD);

    // projections (WMMA)
    dim3 gP(NTOK / 32, PROJ / 128);
    proj_gemm<0><<<gP, 256, 0, stream>>>(Qb, Wq1T, bq1, q1b);
    proj_gemm<0><<<gP, 256, 0, stream>>>(Kb, Wk1T, bk1, k1b);
    proj_gemm<0><<<gP, 256, 0, stream>>>(Qb, Wq2T, bq2, q2b);
    proj_gemm<0><<<gP, 256, 0, stream>>>(Kb, Wk2T, bk2, k2b);
    proj_gemm<1><<<gP, 256, 0, stream>>>(Vb, WvT,  bv,  vvT);   // transposed store

    // two-pass flash attention (WMMA scores + WMMA PV)
    attn_stats<<<NTOK / 16, 256, 0, stream>>>(q1b, k1b, q2b, k2b, m1, l1, m2, l2);
    attn_pv<<<NTOK / 16, 256, 0, stream>>>(q1b, k1b, q2b, k2b, vvT,
                                           m1, l1, m2, l2, scal, attnF);

    // layernorm + output projection (WMMA)
    ln_kernel<<<NTOK, 256, 0, stream>>>(attnF, ln_g, ln_b, scal, attnN);
    out_gemm<<<dim3(NTOK / 32, HIDD / 128), 256, 0, stream>>>(attnN, WoT, bo, out);
}
